// Head_66374424593152
// MI455X (gfx1250) — matrix operand order repaired and checked by reading, not compiled, not run on hardware
//
#include <hip/hip_runtime.h>
#include <hip/hip_bf16.h>
#include <hip/hip_fp16.h>

typedef __attribute__((ext_vector_type(16))) _Float16 v16h;
typedef __attribute__((ext_vector_type(8)))  float    v8f;

#define B_ 32
#define T_ 2048
#define C_ 1024
#define H_ 64

// ---------------------------------------------------------------------------
// Kernel 1: q/k/v projections.  [B*T, C] @ [C, H] for 3 weights.
// Workgroup = 256 threads (8 waves), handles 128 rows. K staged in 32-chunks.
// Output f16 into workspace, order: k, q, v (each B*T*H elements).
// ---------------------------------------------------------------------------
__global__ __launch_bounds__(256)
void proj_kernel(const float* __restrict__ x,
                 const float* __restrict__ Wk,
                 const float* __restrict__ Wq,
                 const float* __restrict__ Wv,
                 _Float16* __restrict__ kqv)
{
    __shared__ _Float16 lx[128 * 32];       // x tile  [row][k]      8 KB
    __shared__ _Float16 lw[3][64 * 32];     // W tiles [n][k] (T)   12 KB

    const int tid  = threadIdx.x;
    const int wave = tid >> 5;
    const int lane = tid & 31;
    const int hh   = lane >> 4;             // lane half (0/1)
    const int nIdx = lane & 15;
    const long R0  = (long)blockIdx.x * 128;

    v8f zero = {};
    v8f acc[3][4];
    #pragma unroll
    for (int a = 0; a < 3; ++a)
        #pragma unroll
        for (int b = 0; b < 4; ++b) acc[a][b] = zero;

    for (int kc = 0; kc < C_ / 32; ++kc) {
        const int kk = kc * 32;
        // stage x (streamed once from HBM -> nontemporal), convert f32->f16
        for (int idx = tid; idx < 128 * 32; idx += 256) {
            int r = idx >> 5, c = idx & 31;
            float xv = __builtin_nontemporal_load(&x[(R0 + r) * C_ + kk + c]);
            lx[idx] = (_Float16)xv;
        }
        // stage W transposed: read W[k][n] coalesced, write LDS[n][k]
        for (int idx = tid; idx < 3 * 32 * 64; idx += 256) {
            int w3  = idx >> 11;
            int rem = idx & 2047;
            int k   = rem >> 6;
            int n   = rem & 63;
            const float* Wp = (w3 == 0) ? Wk : ((w3 == 1) ? Wq : Wv);
            lw[w3][n * 32 + k] = (_Float16)Wp[(kk + k) * H_ + n];
        }
        __syncthreads();

        // A fragment (16x32 f16 layout): lane m = nIdx, K pairs per VGPR
        union { v16h v; unsigned u[8]; } A;
        const unsigned* lxw = (const unsigned*)&lx[(wave * 16 + nIdx) * 32];
        #pragma unroll
        for (int v = 0; v < 8; ++v) {
            int k = hh * 16 + v * 2;
            A.u[v] = lxw[k >> 1];
        }
        #pragma unroll
        for (int w3 = 0; w3 < 3; ++w3) {
            #pragma unroll
            for (int nt = 0; nt < 4; ++nt) {
                // B fragment (32x16 f16): lane n = nIdx, K window by half
                union { v16h v; unsigned u[8]; } Bf;
                const unsigned* lwp =
                    (const unsigned*)&lw[w3][(nt * 16 + nIdx) * 32 + hh * 16];
                #pragma unroll
                for (int v = 0; v < 8; ++v) Bf.u[v] = lwp[v];
                acc[w3][nt] = __builtin_amdgcn_wmma_f32_16x16x32_f16(
                    false, A.v, false, Bf.v, (short)0, acc[w3][nt], false, false);
            }
        }
        __syncthreads();
    }

    // write out (C layout: lanes 0-15 -> M=r, lanes 16-31 -> M=r+8)
    const long BTH = (long)B_ * T_ * H_;
    #pragma unroll
    for (int w3 = 0; w3 < 3; ++w3) {
        _Float16* dst = kqv + (long)w3 * BTH;
        #pragma unroll
        for (int nt = 0; nt < 4; ++nt)
            #pragma unroll
            for (int r = 0; r < 8; ++r) {
                long row = R0 + wave * 16 + r + 8 * hh;
                dst[row * H_ + nt * 16 + nIdx] = (_Float16)acc[w3][nt][r];
            }
    }
}

// ---------------------------------------------------------------------------
// Kernel 2: flash attention. Workgroup = 256 threads (8 waves) = 128 queries
// of one batch; each wave owns a 16-query tile. K block staged via
// GLOBAL_LOAD_ASYNC_TO_LDS_B128 (straight copy, ASYNCcnt-tracked); V block
// transposed through the normal path; online softmax; causal skip.
// ---------------------------------------------------------------------------
__global__ __launch_bounds__(256)
void attn_kernel(const _Float16* __restrict__ kqv,
                 float* __restrict__ out)
{
    __shared__ _Float16 lk[32 * 64];        // K block [key][h]    4 KB
    __shared__ _Float16 lv[64 * 32];        // V block [h][key]    4 KB
    __shared__ _Float16 lp[8][16 * 32];     // per-wave P tile     8 KB

    const int tid  = threadIdx.x;
    const int wave = tid >> 5;
    const int lane = tid & 31;
    const int hh   = lane >> 4;
    const int nIdx = lane & 15;

    const int b    = blockIdx.x >> 4;       // 16 query-blocks per batch
    const int qblk = (blockIdx.x & 15) * 128;
    const int q0   = qblk + wave * 16;

    const long BTH = (long)B_ * T_ * H_;
    const _Float16* kbase = kqv + (long)b * T_ * H_;
    const _Float16* qbase = kqv + BTH + (long)b * T_ * H_;
    const _Float16* vbase = kqv + 2 * BTH + (long)b * T_ * H_;

    // LDS byte offset of lk for the async-to-LDS destination (addrspace(3)
    // flat address low 32 bits == workgroup-relative LDS offset)
    const unsigned lk_off = (unsigned)(size_t)(void*)lk + (unsigned)tid * 16u;

    // Q fragments, A-layout, two 32-wide K chunks over H=64 (kept in VGPRs)
    union { v16h v; unsigned u[8]; } Aq[2];
    {
        const unsigned* qrow = (const unsigned*)(qbase + (long)(q0 + nIdx) * H_);
        #pragma unroll
        for (int c = 0; c < 2; ++c)
            #pragma unroll
            for (int v = 0; v < 8; ++v) {
                int k = c * 32 + hh * 16 + v * 2;
                Aq[c].u[v] = qrow[k >> 1];
            }
    }

    v8f zero = {};
    float rowm[8], rowl[8];
    v8f O[4];
    #pragma unroll
    for (int r = 0; r < 8; ++r) { rowm[r] = -1e30f; rowl[r] = 0.f; }
    #pragma unroll
    for (int nt = 0; nt < 4; ++nt) O[nt] = zero;

    const int kmax = qblk + 127;            // causal bound for whole block
    for (int kb = 0; kb <= kmax; kb += 32) {
        // --- stage K block: 4 KB straight copy via async tensor path ------
        {
            unsigned long long gsrc =
                (unsigned long long)(size_t)(kbase + (long)kb * H_) +
                (unsigned long long)tid * 16ull;
            asm volatile("global_load_async_to_lds_b128 %0, %1, off"
                         :: "v"(lk_off), "v"(gsrc)
                         : "memory");
        }
        // --- stage V transposed: read v[key][h] coalesced, write [h][key] -
        for (int idx = tid; idx < 32 * 64; idx += 256) {
            int key = idx >> 6, h = idx & 63;
            lv[h * 32 + key] = vbase[(long)(kb + key) * H_ + h];
        }
        asm volatile("s_wait_asynccnt 0x0" ::: "memory");
        __syncthreads();

        if (kb <= q0 + 15) {                // wave-uniform causal skip
            // S = Q K^T : 2 key-tiles (n), contraction over H in 2 chunks
            v8f S[2]; S[0] = zero; S[1] = zero;
            #pragma unroll
            for (int t = 0; t < 2; ++t)
                #pragma unroll
                for (int c = 0; c < 2; ++c) {
                    union { v16h v; unsigned u[8]; } Bk;
                    const unsigned* kp =
                        (const unsigned*)&lk[(t * 16 + nIdx) * 64 + c * 32 + hh * 16];
                    #pragma unroll
                    for (int v = 0; v < 8; ++v) Bk.u[v] = kp[v];
                    S[t] = __builtin_amdgcn_wmma_f32_16x16x32_f16(
                        false, Aq[c].v, false, Bk.v, (short)0, S[t], false, false);
                }

            // scale + causal mask + online softmax (rows split across halves)
            const int qrow_ = q0 + 8 * hh;
            float cscale[8];
            #pragma unroll
            for (int r = 0; r < 8; ++r) {
                #pragma unroll
                for (int t = 0; t < 2; ++t) {
                    float s = S[t][r] * 0.125f;               // H^-0.5
                    int key = kb + t * 16 + nIdx;
                    if (key > qrow_ + r) s = -1e30f;          // causal mask
                    S[t][r] = s;
                }
                float mx = fmaxf(S[0][r], S[1][r]);
                #pragma unroll
                for (int off = 8; off >= 1; off >>= 1)
                    mx = fmaxf(mx, __shfl_xor(mx, off, 32));
                float newm = fmaxf(rowm[r], mx);
                float corr = __expf(rowm[r] - newm);
                rowm[r] = newm;
                float p0 = __expf(S[0][r] - newm);
                float p1 = __expf(S[1][r] - newm);
                S[0][r] = p0; S[1][r] = p1;
                float rs = p0 + p1;
                #pragma unroll
                for (int off = 8; off >= 1; off >>= 1)
                    rs += __shfl_xor(rs, off, 32);
                rowl[r] = rowl[r] * corr + rs;
                cscale[r] = corr;
            }
            #pragma unroll
            for (int nt = 0; nt < 4; ++nt)
                #pragma unroll
                for (int r = 0; r < 8; ++r) O[nt][r] *= cscale[r];

            // P: C-layout -> LDS [m][key] -> reload in A-layout (in-order DS)
            _Float16* pp = &lp[wave][0];
            #pragma unroll
            for (int t = 0; t < 2; ++t)
                #pragma unroll
                for (int r = 0; r < 8; ++r)
                    pp[(r + 8 * hh) * 32 + t * 16 + nIdx] = (_Float16)S[t][r];
            union { v16h v; unsigned u[8]; } Ap;
            const unsigned* ppr = (const unsigned*)&pp[nIdx * 32];
            #pragma unroll
            for (int v = 0; v < 8; ++v) {
                int k = hh * 16 + v * 2;
                Ap.u[v] = ppr[k >> 1];
            }

            // O += P @ V  (contraction over 32 keys = one WMMA per h-tile)
            #pragma unroll
            for (int nt = 0; nt < 4; ++nt) {
                union { v16h v; unsigned u[8]; } Bv;
                const unsigned* vp =
                    (const unsigned*)&lv[(nt * 16 + nIdx) * 32 + hh * 16];
                #pragma unroll
                for (int v = 0; v < 8; ++v) Bv.u[v] = vp[v];
                O[nt] = __builtin_amdgcn_wmma_f32_16x16x32_f16(
                    false, Ap.v, false, Bv.v, (short)0, O[nt], false, false);
            }
        }
        __syncthreads();
    }

    // out[b][q][h] = O / l
    #pragma unroll
    for (int r = 0; r < 8; ++r) {
        float inv = 1.0f / rowl[r];
        long q = q0 + r + 8 * hh;
        #pragma unroll
        for (int nt = 0; nt < 4; ++nt)
            out[((long)b * T_ + q) * H_ + nt * 16 + nIdx] = O[nt][r] * inv;
    }
}

extern "C" void kernel_launch(void* const* d_in, const int* in_sizes, int n_in,
                              void* d_out, int out_size, void* d_ws, size_t ws_size,
                              hipStream_t stream) {
    (void)in_sizes; (void)n_in; (void)out_size; (void)ws_size;
    const float* x  = (const float*)d_in[0];
    const float* Wk = (const float*)d_in[1];
    const float* Wq = (const float*)d_in[2];
    const float* Wv = (const float*)d_in[3];
    _Float16* kqv = (_Float16*)d_ws;   // k | q | v, each B*T*H f16 (24 MB)

    proj_kernel<<<(B_ * T_) / 128, 256, 0, stream>>>(x, Wk, Wq, Wv, kqv);
    attn_kernel<<<B_ * (T_ / 128), 256, 0, stream>>>(kqv, (float*)d_out);
}
